// ObjModel_25331717111816
// MI455X (gfx1250) — matrix operand order repaired and checked by reading, not compiled, not run on hardware
//
#include <hip/hip_runtime.h>
#include <hip/hip_bf16.h>
#include <cstdint>

typedef __attribute__((ext_vector_type(16))) _Float16 v16h;
typedef __attribute__((ext_vector_type(8)))  float    v8f;

#define B_N   32
#define S_N   1024
#define H_N   1024
#define BS_N  (B_N * S_N)
#define LDS_STRIDE 40   // padded LDS row stride in f16 elements (80B -> conflict free)

// ---------------- async global->LDS copy (CDNA5, ASYNCcnt tracked) ----------------
// LDS operand is the 32-bit LDS offset = low bits of the flat shared pointer.
__device__ __forceinline__ void async_copy_b128(void* lds_dst, const void* gsrc) {
  uint32_t loff = (uint32_t)(uintptr_t)lds_dst;
  asm volatile("global_load_async_to_lds_b128 %0, %1, off"
               :: "v"(loff), "v"(gsrc)
               : "memory");
}
__device__ __forceinline__ void wait_async0() {
  asm volatile("s_wait_asynccnt 0x0" ::: "memory");
}

// ---------------- block reduction helpers ----------------
__device__ __forceinline__ float block_sum(float v, float* red, int tid) {
  red[tid] = v; __syncthreads();
  #pragma unroll
  for (int s = 128; s > 0; s >>= 1) {
    if (tid < s) red[tid] += red[tid + s];
    __syncthreads();
  }
  float r = red[0]; __syncthreads(); return r;
}
__device__ __forceinline__ float block_max(float v, float* red, int tid) {
  red[tid] = v; __syncthreads();
  #pragma unroll
  for (int s = 128; s > 0; s >>= 1) {
    if (tid < s) red[tid] = fmaxf(red[tid], red[tid + s]);
    __syncthreads();
  }
  float r = red[0]; __syncthreads(); return r;
}

// ---------------- LDS staging ----------------
// A tile: 128 rows x 32 (f16) via async copies (2 x b128 per thread).
__device__ __forceinline__ void stage_A_async(const _Float16* __restrict__ gA, size_t lda,
                                              size_t row0, int k0, _Float16* sA, int tid) {
  int row = tid >> 1;            // 0..127
  int col = (tid & 1) << 4;      // 0 or 16
  const _Float16* src = gA + (row0 + (size_t)row) * lda + (size_t)(k0 + col);
  _Float16* dst = sA + row * LDS_STRIDE + col;
  async_copy_b128(dst, src);
  async_copy_b128(dst + 8, src + 8);
}
// B tile: 64 n-rows x 32 k (f16), source row-major [n][k]; 1 x b128 per thread.
__device__ __forceinline__ void stage_B_nk_async(const _Float16* __restrict__ gB, size_t ldb,
                                                 size_t n0, int k0, _Float16* sB, int tid) {
  int n   = tid >> 2;            // 0..63
  int col = (tid & 3) << 3;      // 0,8,16,24
  async_copy_b128(sB + n * LDS_STRIDE + col,
                  gB + (n0 + (size_t)n) * ldb + (size_t)(k0 + col));
}
// B tile from [k][n] source (transpose during staging): sB[n][k] = gB[k0+k][n0+n]
__device__ __forceinline__ void stage_B_kn(const _Float16* __restrict__ gB, size_t ldb,
                                           int k0, size_t n0, _Float16* sB, int tid) {
  int kk = tid >> 3;             // 0..31
  int nb = (tid & 7) << 3;       // 0..56
  union { uint4 v; _Float16 h[8]; } u;
  u.v = *reinterpret_cast<const uint4*>(gB + (size_t)(k0 + kk) * ldb + n0 + (size_t)nb);
  #pragma unroll
  for (int j = 0; j < 8; ++j) sB[(nb + j) * LDS_STRIDE + kk] = u.h[j];
}

// ---------------- WMMA fragment builders (per ISA 16-bit layouts) ----------------
// A 16x32: lane(0-15)=row M, K pairs V0..V3:{0,1..6,7}+8*hf, V4..V7:{16..23}+8*hf
__device__ __forceinline__ v16h frag_A(const _Float16* sA, int mbase, int r, int hf) {
  const _Float16* row = sA + (mbase + r) * LDS_STRIDE;
  v16h a;
  #pragma unroll
  for (int p = 0; p < 8; ++p) {
    int kb = ((p < 4) ? (p * 2) : (16 + (p - 4) * 2)) + hf * 8;
    a[2 * p]     = row[kb];
    a[2 * p + 1] = row[kb + 1];
  }
  return a;
}
// B 32x16: lane col N, K {0..7,16..23}+8*hf (same order as A)
__device__ __forceinline__ v16h frag_B(const _Float16* sB, int nbase, int r, int hf) {
  const _Float16* row = sB + (nbase + r) * LDS_STRIDE + hf * 8;
  v16h b;
  #pragma unroll
  for (int i = 0; i < 16; ++i) b[i] = row[(i < 8) ? i : (i + 8)];
  return b;
}

// one K=32 step for a wave's 32x32 tile (2x2 of 16x16 WMMA)
__device__ __forceinline__ void mma_step(const _Float16* sA, const _Float16* sB,
                                         int wm, int wn, int r, int hf, v8f (&acc)[2][2]) {
  v16h af[2], bf[2];
  #pragma unroll
  for (int mi = 0; mi < 2; ++mi) af[mi] = frag_A(sA, wm * 32 + mi * 16, r, hf);
  #pragma unroll
  for (int ni = 0; ni < 2; ++ni) bf[ni] = frag_B(sB, wn * 32 + ni * 16, r, hf);
  #pragma unroll
  for (int mi = 0; mi < 2; ++mi)
    #pragma unroll
    for (int ni = 0; ni < 2; ++ni)
      acc[mi][ni] = __builtin_amdgcn_wmma_f32_16x16x32_f16(
          false, af[mi], false, bf[ni], (short)0, acc[mi][ni], false, false);
}

// ---------------- kernel 0: convert Wq/Wk/Wv to f16 ----------------
__global__ __launch_bounds__(256) void convert_w_kernel(const float* __restrict__ Wq,
                                                        const float* __restrict__ Wk,
                                                        const float* __restrict__ Wv,
                                                        _Float16* __restrict__ Wh) {
  size_t i = (size_t)blockIdx.x * 256 + threadIdx.x;   // grid covers 3*H*H exactly
  size_t hh = (size_t)H_N * H_N;
  int w = (int)(i / hh);
  size_t rme = i % hh;
  const float* p = (w == 0) ? Wq : (w == 1) ? Wk : Wv;
  Wh[i] = (_Float16)p[rme];
}

// ---------------- kernel 1: per-batch cond-LN params ----------------
__global__ __launch_bounds__(256) void prep_kernel(
    const int* __restrict__ relation, const float* __restrict__ x,
    const int* __restrict__ sub_head, const int* __restrict__ sub_tail,
    const float* __restrict__ rel_emb, const float* __restrict__ Wrf,
    const float* __restrict__ brf, const float* __restrict__ cln_w,
    const float* __restrict__ cln_b, const float* __restrict__ Wwd,
    const float* __restrict__ Wbd, float* __restrict__ wvec, float* __restrict__ bvec) {
  __shared__ float s_in[H_N];
  __shared__ float s_rel[H_N];
  __shared__ float s_sub[H_N];
  const int b = blockIdx.x;
  const int tid = threadIdx.x;
  const int relid = relation[b];
  const int sh = sub_head[b];
  const int st = sub_tail[b];
  #pragma unroll
  for (int c = 0; c < 4; ++c) s_in[tid + c * 256] = rel_emb[(size_t)relid * H_N + tid + c * 256];
  __syncthreads();
  #pragma unroll
  for (int c = 0; c < 4; ++c) {
    int g = tid + c * 256;
    float acc = brf[g];
    const float* wr = Wrf + (size_t)g * H_N;
    for (int h = 0; h < H_N; ++h) acc += s_in[h] * wr[h];
    s_rel[g] = fminf(fmaxf(acc, 0.0f), 6.0f);      // ReLU6
  }
  #pragma unroll
  for (int c = 0; c < 4; ++c) {
    int g = tid + c * 256;
    s_sub[g] = 0.5f * (x[((size_t)b * S_N + sh) * H_N + g] + x[((size_t)b * S_N + st) * H_N + g]);
  }
  __syncthreads();
  #pragma unroll
  for (int c = 0; c < 4; ++c) {
    int g = tid + c * 256;
    const float* ww = Wwd + (size_t)g * (2 * H_N);
    const float* wb = Wbd + (size_t)g * (2 * H_N);
    float aw = cln_w[g], ab = cln_b[g];
    for (int h = 0; h < H_N; ++h) {
      float rr = s_rel[h], ss = s_sub[h];
      aw += rr * ww[h] + ss * ww[H_N + h];
      ab += rr * wb[h] + ss * wb[H_N + h];
    }
    wvec[(size_t)b * H_N + g] = aw;
    bvec[(size_t)b * H_N + g] = ab;
  }
}

// ---------------- kernel 2: conditional LayerNorm ----------------
__global__ __launch_bounds__(256) void ln_kernel(const float* __restrict__ x,
                                                 const float* __restrict__ wvec,
                                                 const float* __restrict__ bvec,
                                                 float* __restrict__ hidden,
                                                 _Float16* __restrict__ normed_h) {
  __shared__ float red[256];
  const int t = blockIdx.x;
  const int b = t / S_N;
  const int tid = threadIdx.x;
  const float* xr = x + (size_t)t * H_N;
  float4 xv = *reinterpret_cast<const float4*>(xr + tid * 4);
  float mean = block_sum(xv.x + xv.y + xv.z + xv.w, red, tid) * (1.0f / H_N);
  float4 c = make_float4(xv.x - mean, xv.y - mean, xv.z - mean, xv.w - mean);
  float var = block_sum(c.x * c.x + c.y * c.y + c.z * c.z + c.w * c.w, red, tid) * (1.0f / H_N);
  float inv = rsqrtf(var + 1e-12f);
  const float* wv = wvec + (size_t)b * H_N + tid * 4;
  const float* bv = bvec + (size_t)b * H_N + tid * 4;
  float o0 = c.x * inv * wv[0] + bv[0];
  float o1 = c.y * inv * wv[1] + bv[1];
  float o2 = c.z * inv * wv[2] + bv[2];
  float o3 = c.w * inv * wv[3] + bv[3];
  *reinterpret_cast<float4*>(hidden + (size_t)t * H_N + tid * 4) = make_float4(o0, o1, o2, o3);
  _Float16* nh = normed_h + (size_t)t * H_N + tid * 4;
  nh[0] = (_Float16)o0; nh[1] = (_Float16)o1; nh[2] = (_Float16)o2; nh[3] = (_Float16)o3;
}

// ---------------- kernel 3: QKV WMMA GEMM (relu(normed @ W^T + b) -> f16) ----------------
__global__ __launch_bounds__(256) void qkv_gemm_kernel(
    const _Float16* __restrict__ normed_h, const _Float16* __restrict__ Wh,
    const float* __restrict__ bq, const float* __restrict__ bk, const float* __restrict__ bv,
    _Float16* __restrict__ qkv) {
  __shared__ alignas(16) _Float16 As[128 * LDS_STRIDE];
  __shared__ alignas(16) _Float16 Bs[64 * LDS_STRIDE];
  const int tid = threadIdx.x;
  const int w = blockIdx.z;
  const size_t t0 = (size_t)blockIdx.x * 128;
  const size_t g0 = (size_t)blockIdx.y * 64;
  const _Float16* gB = Wh + (size_t)w * H_N * H_N;
  const int wid = tid >> 5, lane = tid & 31;
  const int wm = wid >> 1, wn = wid & 1;
  const int hf = lane >> 4, r = lane & 15;
  v8f acc[2][2] = {};
  for (int k0 = 0; k0 < H_N; k0 += 32) {
    __syncthreads();
    stage_A_async(normed_h, H_N, t0, k0, As, tid);
    stage_B_nk_async(gB, H_N, g0, k0, Bs, tid);
    if (k0 + 32 < H_N)
      __builtin_prefetch(normed_h + (t0 + (size_t)(tid >> 1)) * H_N + k0 + 32, 0, 0);
    wait_async0();               // this wave's async LDS writes have landed
    __syncthreads();             // all waves' tiles visible
    mma_step(As, Bs, wm, wn, r, hf, acc);
  }
  const float* bias = (w == 0) ? bq : (w == 1) ? bk : bv;
  _Float16* out = qkv + (size_t)w * BS_N * H_N;
  #pragma unroll
  for (int mi = 0; mi < 2; ++mi) {
    #pragma unroll
    for (int ni = 0; ni < 2; ++ni) {
      size_t gcol = g0 + wn * 32 + ni * 16 + r;
      float bb = bias[gcol];
      size_t rowb = t0 + wm * 32 + mi * 16 + hf * 8;
      #pragma unroll
      for (int j = 0; j < 8; ++j) {
        float vv = fmaxf(acc[mi][ni][j] + bb, 0.0f);
        out[(rowb + j) * H_N + gcol] = (_Float16)vv;
      }
    }
  }
}

// ---------------- kernel 4: attention scores e = q k^T (f32) ----------------
__global__ __launch_bounds__(256) void scores_gemm_kernel(const _Float16* __restrict__ qkv,
                                                          float* __restrict__ e) {
  __shared__ alignas(16) _Float16 As[128 * LDS_STRIDE];
  __shared__ alignas(16) _Float16 Bs[64 * LDS_STRIDE];
  const int tid = threadIdx.x;
  const int b = blockIdx.z;
  const size_t i0 = (size_t)blockIdx.x * 128;
  const size_t j0 = (size_t)blockIdx.y * 64;
  const _Float16* gA = qkv;                                   // q
  const _Float16* gB = qkv + (size_t)1 * BS_N * H_N;          // k
  const int wid = tid >> 5, lane = tid & 31;
  const int wm = wid >> 1, wn = wid & 1;
  const int hf = lane >> 4, r = lane & 15;
  v8f acc[2][2] = {};
  for (int k0 = 0; k0 < H_N; k0 += 32) {
    __syncthreads();
    stage_A_async(gA, H_N, (size_t)b * S_N + i0, k0, As, tid);
    stage_B_nk_async(gB, H_N, (size_t)b * S_N + j0, k0, Bs, tid);
    wait_async0();
    __syncthreads();
    mma_step(As, Bs, wm, wn, r, hf, acc);
  }
  #pragma unroll
  for (int mi = 0; mi < 2; ++mi) {
    #pragma unroll
    for (int ni = 0; ni < 2; ++ni) {
      size_t col = j0 + wn * 32 + ni * 16 + r;
      size_t rowb = (size_t)b * S_N + i0 + wm * 32 + mi * 16 + hf * 8;
      #pragma unroll
      for (int j = 0; j < 8; ++j) e[(rowb + j) * S_N + col] = acc[mi][ni][j];
    }
  }
}

// ---------------- kernel 5: masked softmax row-wise, f32 e -> f16 p in place ----------------
__global__ __launch_bounds__(256) void softmax_kernel(float* __restrict__ e,
                                                      const int* __restrict__ mask) {
  __shared__ float red[256];
  const int t = blockIdx.x;          // global row index b*S + i
  const int b = t / S_N;
  const int tid = threadIdx.x;
  float* erow = e + (size_t)t * S_N;
  const int* mrow = mask + (size_t)b * S_N;
  int j0 = tid * 4;
  float4 ev = *reinterpret_cast<const float4*>(erow + j0);
  float vals[4] = {ev.x, ev.y, ev.z, ev.w};
  #pragma unroll
  for (int j = 0; j < 4; ++j) {
    float mm = (float)mrow[j0 + j];
    vals[j] -= 1e10f * (1.0f - mm);
  }
  float lm = fmaxf(fmaxf(vals[0], vals[1]), fmaxf(vals[2], vals[3]));
  float gmax = block_max(lm, red, tid);
  float ls = 0.0f;
  #pragma unroll
  for (int j = 0; j < 4; ++j) { vals[j] = __expf(vals[j] - gmax); ls += vals[j]; }
  float gsum = block_sum(ls, red, tid);       // ends with a barrier: all reads done
  float inv = 1.0f / gsum;
  _Float16* prow = reinterpret_cast<_Float16*>(erow);
  #pragma unroll
  for (int j = 0; j < 4; ++j) prow[j0 + j] = (_Float16)(vals[j] * inv);
}

// ---------------- kernel 6: hidden += p @ v (WMMA, residual add) ----------------
__global__ __launch_bounds__(256) void av_gemm_kernel(const float* __restrict__ e_as_p,
                                                      const _Float16* __restrict__ qkv,
                                                      float* __restrict__ hidden) {
  __shared__ alignas(16) _Float16 As[128 * LDS_STRIDE];
  __shared__ alignas(16) _Float16 Bs[64 * LDS_STRIDE];
  const int tid = threadIdx.x;
  const int b = blockIdx.z;
  const size_t i0 = (size_t)blockIdx.x * 128;
  const size_t g0 = (size_t)blockIdx.y * 64;
  const _Float16* gA = reinterpret_cast<const _Float16*>(e_as_p);   // f16 probs, row stride 2*S
  const _Float16* gB = qkv + (size_t)2 * BS_N * H_N;                // v, [token][h] row-major
  const int wid = tid >> 5, lane = tid & 31;
  const int wm = wid >> 1, wn = wid & 1;
  const int hf = lane >> 4, r = lane & 15;
  v8f acc[2][2] = {};
  for (int k0 = 0; k0 < S_N; k0 += 32) {
    __syncthreads();
    stage_A_async(gA, (size_t)2 * S_N, (size_t)b * S_N + i0, k0, As, tid);  // async probs tile
    stage_B_kn(gB, H_N, b * S_N + k0, g0, Bs, tid);                 // transpose v tile via VGPRs
    wait_async0();
    __syncthreads();
    mma_step(As, Bs, wm, wn, r, hf, acc);
  }
  #pragma unroll
  for (int mi = 0; mi < 2; ++mi) {
    #pragma unroll
    for (int ni = 0; ni < 2; ++ni) {
      size_t gcol = g0 + wn * 32 + ni * 16 + r;
      size_t rowb = (size_t)b * S_N + i0 + wm * 32 + mi * 16 + hf * 8;
      #pragma unroll
      for (int j = 0; j < 8; ++j) {
        float* hp = hidden + (rowb + j) * H_N + gcol;
        *hp = *hp + acc[mi][ni][j];      // residual: hidden already holds normed
      }
    }
  }
}

// ---------------- kernel 7: pred = hidden @ Wo^T + bo ----------------
__global__ __launch_bounds__(256) void pred_kernel(const float* __restrict__ hidden,
                                                   const float* __restrict__ Wo,
                                                   const float* __restrict__ bo,
                                                   float* __restrict__ pred) {
  __shared__ float red[256];
  const int t = blockIdx.x;
  const int tid = threadIdx.x;
  const float* hr = hidden + (size_t)t * H_N;
  float4 hv = *reinterpret_cast<const float4*>(hr + tid * 4);
  float4 wv = *reinterpret_cast<const float4*>(Wo + tid * 4);
  float s = hv.x * wv.x + hv.y * wv.y + hv.z * wv.z + hv.w * wv.w;
  float tot = block_sum(s, red, tid);
  if (tid == 0) pred[t] = tot + bo[0];
}

// ---------------- launch ----------------
extern "C" void kernel_launch(void* const* d_in, const int* in_sizes, int n_in,
                              void* d_out, int out_size, void* d_ws, size_t ws_size,
                              hipStream_t stream) {
  const int*   relation = (const int*)  d_in[0];
  const float* x        = (const float*)d_in[1];
  const int*   sub_head = (const int*)  d_in[2];
  const int*   sub_tail = (const int*)  d_in[3];
  const int*   att_mask = (const int*)  d_in[4];
  const float* rel_emb  = (const float*)d_in[5];
  const float* Wrf      = (const float*)d_in[6];
  const float* brf      = (const float*)d_in[7];
  const float* cln_w    = (const float*)d_in[8];
  const float* cln_b    = (const float*)d_in[9];
  const float* Wwd      = (const float*)d_in[10];
  const float* Wbd      = (const float*)d_in[11];
  const float* Wq       = (const float*)d_in[12];
  const float* bq       = (const float*)d_in[13];
  const float* Wk       = (const float*)d_in[14];
  const float* bk       = (const float*)d_in[15];
  const float* Wv       = (const float*)d_in[16];
  const float* bv       = (const float*)d_in[17];
  const float* Wo       = (const float*)d_in[18];
  const float* bo       = (const float*)d_in[19];

  char* ws = (char*)d_ws;
  size_t off = 0;
  _Float16* normed_h = (_Float16*)(ws + off); off += (size_t)BS_N * H_N * 2;       // 64 MiB
  _Float16* qkv      = (_Float16*)(ws + off); off += (size_t)3 * BS_N * H_N * 2;   // 192 MiB
  _Float16* Wh       = (_Float16*)(ws + off); off += (size_t)3 * H_N * H_N * 2;    // 6 MiB
  float*    wvec     = (float*)   (ws + off); off += (size_t)B_N * H_N * 4;
  float*    bvec     = (float*)   (ws + off); off += (size_t)B_N * H_N * 4;
  float*    e        = (float*)   (ws + off); off += (size_t)B_N * S_N * S_N * 4;  // 128 MiB

  float* pred   = (float*)d_out;
  float* hidden = (float*)d_out + BS_N;

  convert_w_kernel<<<dim3(3 * H_N * H_N / 256), dim3(256), 0, stream>>>(Wq, Wk, Wv, Wh);
  prep_kernel<<<dim3(B_N), dim3(256), 0, stream>>>(relation, x, sub_head, sub_tail, rel_emb,
                                                   Wrf, brf, cln_w, cln_b, Wwd, Wbd, wvec, bvec);
  ln_kernel<<<dim3(BS_N), dim3(256), 0, stream>>>(x, wvec, bvec, hidden, normed_h);
  qkv_gemm_kernel<<<dim3(BS_N / 128, H_N / 64, 3), dim3(256), 0, stream>>>(
      normed_h, Wh, bq, bk, bv, qkv);
  scores_gemm_kernel<<<dim3(S_N / 128, S_N / 64, B_N), dim3(256), 0, stream>>>(qkv, e);
  softmax_kernel<<<dim3(BS_N), dim3(256), 0, stream>>>(e, att_mask);
  av_gemm_kernel<<<dim3(S_N / 128, H_N / 64, B_N), dim3(256), 0, stream>>>(e, qkv, hidden);
  pred_kernel<<<dim3(BS_N), dim3(256), 0, stream>>>(hidden, Wo, bo, pred);
}
